// EuclidConv_75265006895560
// MI455X (gfx1250) — matrix operand order repaired and checked by reading, not compiled, not run on hardware
//
#include <hip/hip_runtime.h>

// ---------------- types ----------------
typedef __attribute__((ext_vector_type(16))) __bf16   v16bf;
typedef __attribute__((ext_vector_type(8)))  float    v8f;
typedef __attribute__((ext_vector_type(4)))  unsigned u32x4;

union Frag {
    v16bf    v;     // WMMA operand (8 VGPRs, 16 bf16)
    unsigned u[8];  // dword view: u[i] = {K lo half, K hi half} packed bf16
    u32x4    q[2];  // 128-bit load view
};

// Problem constants
#define CIN   128
#define COUT  256
#define HW    784          // 28*28
#define NIMG  32
#define NPIX  25088        // 32*784
#define KTOT  1152         // 128*9
#define NSTEP 36           // 1152 / 32
#define XSTRIDE (CIN*HW)   // 100352
#define OSTRIDE (COUT*HW)  // 200704

// ---------------- bf16 helpers ----------------
__device__ __forceinline__ unsigned f32_to_bf16_rne(float f) {
    unsigned u = __float_as_uint(f);
    unsigned r = u + 0x7FFFu + ((u >> 16) & 1u);
    return r >> 16;
}
__device__ __forceinline__ float bf16_bits_to_f32(unsigned b) {
    return __uint_as_float(b << 16);
}

// ---------------------------------------------------------------------------
// 1) Pre-swizzle weights into WMMA A-fragment layout, bf16 hi/lo split.
//    A (16-bit, 16x32): lane L holds row M = L%16; dword v, half h ->
//      K = 16*(L>>4) + 2*v + h
//    frag index = ((tile*36 + kstep)*32 + lane)*8 + v
//    kstep = (kh*3+kw)*4 + cichunk ; K within chunk selects ci offset.
// ---------------------------------------------------------------------------
__global__ void prep_weights(const float* __restrict__ w,
                             unsigned* __restrict__ whi,
                             unsigned* __restrict__ wlo) {
    int idx = blockIdx.x * blockDim.x + threadIdx.x;   // 16*36*32*8 = 147456
    if (idx >= 16 * NSTEP * 32 * 8) return;
    int v     = idx & 7;
    int lane  = (idx >> 3) & 31;
    int kstep = (idx >> 8) % NSTEP;
    int tile  = idx / (NSTEP * 256);

    int cout  = tile * 16 + (lane & 15);
    int kbase = 2 * v + ((lane >> 4) << 4);
    int pos = kstep >> 2, cic = kstep & 3;
    int kh = pos / 3, kw = pos - 3 * kh;

    unsigned hi = 0, lo = 0;
    #pragma unroll
    for (int h = 0; h < 2; ++h) {
        int ci = cic * 32 + kbase + h;
        float f = w[cout * KTOT + ci * 9 + kh * 3 + kw];
        unsigned hb = f32_to_bf16_rne(f);
        float rem   = f - bf16_bits_to_f32(hb);
        unsigned lb = f32_to_bf16_rne(rem);
        hi |= hb << (16 * h);
        lo |= lb << (16 * h);
    }
    whi[idx] = hi;
    wlo[idx] = lo;
}

// ---------------------------------------------------------------------------
// 2) t2[co] = sum(w[co]^2)  (LDS tree reduction)
// ---------------------------------------------------------------------------
__global__ void filter_norms(const float* __restrict__ w, float* __restrict__ t2) {
    __shared__ float sm[128];
    int cout = blockIdx.x;
    int tid  = threadIdx.x;                 // 128 threads, one per ci
    const float* wp = w + cout * KTOT + tid * 9;
    float a = 0.f;
    #pragma unroll
    for (int i = 0; i < 9; ++i) { float v = wp[i]; a += v * v; }
    sm[tid] = a;
    __syncthreads();
    for (int off = 64; off > 0; off >>= 1) {
        if (tid < off) sm[tid] += sm[tid + off];
        __syncthreads();
    }
    if (tid == 0) t2[cout] = sm[0];
}

// ---------------------------------------------------------------------------
// 3) t1[p] = sum over 3x3xCIN patch of x^2 (zero-padded)
// ---------------------------------------------------------------------------
__global__ void t1_kernel(const float* __restrict__ x, float* __restrict__ t1) {
    int p = blockIdx.x * blockDim.x + threadIdx.x;
    if (p >= NPIX) return;
    int n = p / HW, s = p % HW;
    int oh = s / 28, ow = s % 28;
    const float* xn = x + n * XSTRIDE;
    float acc = 0.f;
    for (int kh = 0; kh < 3; ++kh) {
        int ih = oh + kh - 1;
        if ((unsigned)ih >= 28u) continue;
        for (int kw = 0; kw < 3; ++kw) {
            int iw = ow + kw - 1;
            if ((unsigned)iw >= 28u) continue;
            const float* xp = xn + ih * 28 + iw;
            for (int ci = 0; ci < CIN; ++ci) {
                float v = xp[ci * HW];
                acc += v * v;
            }
        }
    }
    t1[p] = acc;
}

// ---------------------------------------------------------------------------
// 4) Implicit-GEMM conv via v_wmma_f32_16x16x32_bf16 with bf16x3 split.
//    Block = 128 threads (4 waves). Block -> one 16-pixel tile; wave w ->
//    couts [64w, 64w+64) = 4 WMMA row-tiles, so each B-fragment build is
//    amortized over 12 WMMA ops. D layout: lane holds column
//    N = pixel = lane%16; row M(co) = j + 8*(lane/16) per acc dword j.
//    B (dense 32x16 bf16): lane = col N; dword v half h -> K = 16*(L>>4)+2v+h.
// ---------------------------------------------------------------------------
__global__ __launch_bounds__(128) void conv_wmma(
        const float*    __restrict__ x,
        const unsigned* __restrict__ whi,
        const unsigned* __restrict__ wlo,
        const float*    __restrict__ t1,
        const float*    __restrict__ t2,
        float*          __restrict__ out) {
    const int lane  = threadIdx.x & 31;
    const int wave  = threadIdx.x >> 5;
    const int p0    = blockIdx.x * 16;
    const int cout0 = wave * 64;          // 4 waves x 64 couts = 256
    const int tile0 = cout0 >> 4;

    // this lane's pixel (B/D column)
    const int p  = p0 + (lane & 15);
    const int n  = p / HW;
    const int s  = p % HW;
    const int oh = s / 28, ow = s % 28;
    const float* xn = x + n * XSTRIDE;

    const int chalf = (lane >> 4) << 4;   // +16 channels for upper half-wave

    v8f acc[4];
    acc[0] = v8f{};
    acc[1] = v8f{};
    acc[2] = v8f{};
    acc[3] = v8f{};

    for (int kstep = 0; kstep < NSTEP; ++kstep) {
        const int pos = kstep >> 2, cic = kstep & 3;
        const int kh = pos / 3, kw = pos - 3 * kh;
        const int ih = oh + kh - 1, iw = ow + kw - 1;
        const bool valid = ((unsigned)ih < 28u) && ((unsigned)iw < 28u);

        // ---- build B fragments (x side), hi and lo bf16 ----
        Frag bhi, blo;
        const float* xp = xn + (cic * 32 + chalf) * HW + ih * 28 + iw;
        #pragma unroll
        for (int v = 0; v < 8; ++v) {
            float f0 = valid ? xp[(2 * v)     * HW] : 0.f;
            float f1 = valid ? xp[(2 * v + 1) * HW] : 0.f;
            unsigned h0 = f32_to_bf16_rne(f0);
            unsigned h1 = f32_to_bf16_rne(f1);
            float r0 = f0 - bf16_bits_to_f32(h0);
            float r1 = f1 - bf16_bits_to_f32(h1);
            bhi.u[v] = h0 | (h1 << 16);
            blo.u[v] = f32_to_bf16_rne(r0) | (f32_to_bf16_rne(r1) << 16);
        }

        // prefetch next k-step's weight fragments into cache
        if (kstep + 1 < NSTEP) {
            unsigned noff = (((unsigned)(tile0 * NSTEP + kstep + 1) * 32u + lane) << 3);
            __builtin_prefetch(whi + noff, 0, 0);
            __builtin_prefetch(wlo + noff, 0, 0);
        }

        // ---- four cout tiles share one B build ----
        #pragma unroll
        for (int t = 0; t < 4; ++t) {
            const unsigned base =
                (((unsigned)((tile0 + t) * NSTEP + kstep) * 32u + lane) << 3);
            Frag ahi, alo;
            const u32x4* ph = (const u32x4*)(whi + base);
            const u32x4* pl = (const u32x4*)(wlo + base);
            ahi.q[0] = ph[0]; ahi.q[1] = ph[1];
            alo.q[0] = pl[0]; alo.q[1] = pl[1];

            // bf16x3: hi*hi + hi*lo + lo*hi  (f32 accumulate)
            acc[t] = __builtin_amdgcn_wmma_f32_16x16x32_bf16(
                         false, ahi.v, false, bhi.v, (short)0, acc[t], false, false);
            acc[t] = __builtin_amdgcn_wmma_f32_16x16x32_bf16(
                         false, ahi.v, false, blo.v, (short)0, acc[t], false, false);
            acc[t] = __builtin_amdgcn_wmma_f32_16x16x32_bf16(
                         false, alo.v, false, bhi.v, (short)0, acc[t], false, false);
        }
    }

    // ---- epilogue: out = 2*conv + t1[p] + t2[co] ----
    const float t1v = t1[p];
    float* ob = out + n * OSTRIDE + s;
    #pragma unroll
    for (int t = 0; t < 4; ++t) {
        #pragma unroll
        for (int j = 0; j < 8; ++j) {
            int cout = cout0 + 16 * t + j + 8 * (lane >> 4);
            ob[cout * HW] = 2.f * acc[t][j] + t1v + t2[cout];
        }
    }
}

// ---------------------------------------------------------------------------
// 5) Per-channel batch stats (deterministic fixed-order reduction)
// ---------------------------------------------------------------------------
__global__ void stats_kernel(const float* __restrict__ out,
                             float* __restrict__ mean,
                             float* __restrict__ invstd) {
    __shared__ float ssum[256];
    __shared__ float ssq[256];
    const int c = blockIdx.x;
    const int tid = threadIdx.x;
    float sum = 0.f, sq = 0.f;
    for (int i = tid; i < NPIX; i += 256) {
        int nimg = i / HW, sidx = i % HW;
        float v = out[nimg * OSTRIDE + c * HW + sidx];
        sum += v;
        sq  += v * v;
    }
    ssum[tid] = sum; ssq[tid] = sq;
    __syncthreads();
    for (int off = 128; off > 0; off >>= 1) {
        if (tid < off) { ssum[tid] += ssum[tid + off]; ssq[tid] += ssq[tid + off]; }
        __syncthreads();
    }
    if (tid == 0) {
        float m   = ssum[0] * (1.f / NPIX);
        float var = ssq[0] * (1.f / NPIX) - m * m;   // biased, as torch/jnp.var
        mean[c]   = m;
        invstd[c] = rsqrtf(var + 1e-5f);
    }
}

// ---------------------------------------------------------------------------
// 6) Apply BN affine
// ---------------------------------------------------------------------------
__global__ void norm_kernel(float* __restrict__ out,
                            const float* __restrict__ mean,
                            const float* __restrict__ invstd,
                            const float* __restrict__ gamma,
                            const float* __restrict__ beta) {
    int idx = blockIdx.x * blockDim.x + threadIdx.x;
    if (idx >= NIMG * OSTRIDE) return;
    int c = (idx / HW) & (COUT - 1);
    out[idx] = (out[idx] - mean[c]) * invstd[c] * gamma[c] + beta[c];
}

// ---------------------------------------------------------------------------
extern "C" void kernel_launch(void* const* d_in, const int* in_sizes, int n_in,
                              void* d_out, int out_size, void* d_ws, size_t ws_size,
                              hipStream_t stream) {
    const float* x     = (const float*)d_in[0];
    const float* w     = (const float*)d_in[1];
    const float* gamma = (const float*)d_in[2];
    const float* beta  = (const float*)d_in[3];
    float* out = (float*)d_out;

    // workspace layout (dwords): whi 147456 | wlo 147456 | t1 25088 | t2 256
    //                            mean 256 | invstd 256   (~1.28 MB total)
    unsigned* whi   = (unsigned*)d_ws;
    unsigned* wlo   = whi + 16 * NSTEP * 32 * 8;
    float*    t1    = (float*)(wlo + 16 * NSTEP * 32 * 8);
    float*    t2    = t1 + NPIX;
    float*    meanb = t2 + COUT;
    float*    istdb = meanb + COUT;

    prep_weights<<<576, 256, 0, stream>>>(w, whi, wlo);
    filter_norms<<<COUT, 128, 0, stream>>>(w, t2);
    t1_kernel<<<NPIX / 128, 128, 0, stream>>>(x, t1);
    conv_wmma<<<NPIX / 16, 128, 0, stream>>>(x, whi, wlo, t1, t2, out);
    stats_kernel<<<COUT, 256, 0, stream>>>(out, meanb, istdb);
    norm_kernel<<<(NIMG * OSTRIDE) / 256, 256, 0, stream>>>(out, meanb, istdb, gamma, beta);
}
